// GapLoss_7868380086297
// MI455X (gfx1250) — matrix operand order repaired and checked by reading, not compiled, not run on hardware
//
#include <hip/hip_runtime.h>
#include <hip/hip_bf16.h>

// ---------------- problem constants ----------------
constexpr int BATCH = 8, NCLS = 2, H = 512, W = 512;
constexpr int HW   = H * W;            // 262144 = 1<<18
constexpr int NPIX = BATCH * HW;       // 2097152
constexpr int TILES = BATCH * (H / 16) * (W / 16); // 8192 16x16 tiles
constexpr int ITERS = 64;              // fixed upper bound on Zhang-Suen body iterations
constexpr float KSCALE = 60.0f;

// ---------------- WMMA vector types (CDNA5, wave32) ----------------
typedef __attribute__((ext_vector_type(16))) __bf16 v16bf;
typedef __attribute__((ext_vector_type(8)))  float  v8f;

// ---------------- helpers ----------------
__device__ __forceinline__ int getpix(const unsigned char* img, int y, int x) {
    return (y >= 0 && y < H && x >= 0 && x < W) ? (int)img[(y << 9) | x] : 0;
}

// ---------------- 1) per-pixel CE loss + argmax mask + flag reset ----------------
__global__ void prep_kernel(const float* __restrict__ pred, const int* __restrict__ target,
                            float* __restrict__ Lbuf, unsigned char* __restrict__ xbuf,
                            int* __restrict__ flags) {
    int gid = blockIdx.x * blockDim.x + threadIdx.x;     // exact: NPIX threads
    int b   = gid >> 18;
    int rem = gid & (HW - 1);
    float p0 = pred[(b << 19) + rem];          // b*2*HW + 0*HW
    float p1 = pred[(b << 19) + HW + rem];     // class 1
    int   t  = target[gid];
    float mx  = fmaxf(p0, p1);
    float lse = mx + log1pf(expf(-fabsf(p0 - p1)));      // logsumexp, 2 classes
    float pt  = t ? p1 : p0;
    Lbuf[gid] = lse - pt;                                 // -log_softmax[target]
    xbuf[gid] = (p1 > p0) ? 1 : 0;                        // argmax != 0 (tie -> class 0)
    if (gid == 0) { flags[0] = 0; flags[1] = 0; }         // done, changed
}

// ---------------- 2) one Zhang-Suen substep (early-out on convergence) ----------------
__global__ void thin_kernel(const unsigned char* __restrict__ in, unsigned char* __restrict__ out,
                            int* __restrict__ flags, int firstPass) {
    __shared__ int sdone, schanged;
    if (threadIdx.x == 0) { sdone = flags[0]; schanged = 0; }
    __syncthreads();
    if (sdone) return;                                    // converged: skip (uniform exit)

    int gid = blockIdx.x * blockDim.x + threadIdx.x;
    int b   = gid >> 18;
    int rem = gid & (HW - 1);
    int y = rem >> 9, x = rem & 511;
    const unsigned char* img = in + (b << 18);
    int c = img[rem];
    int del = 0;
    if (c) {
        // P2..P9 (N, NE, E, SE, S, SW, W, NW) with zero padding
        int n0 = getpix(img, y - 1, x    );
        int n1 = getpix(img, y - 1, x + 1);
        int n2 = getpix(img, y,     x + 1);
        int n3 = getpix(img, y + 1, x + 1);
        int n4 = getpix(img, y + 1, x    );
        int n5 = getpix(img, y + 1, x - 1);
        int n6 = getpix(img, y,     x - 1);
        int n7 = getpix(img, y - 1, x - 1);
        int Bn = n0 + n1 + n2 + n3 + n4 + n5 + n6 + n7;
        int A  = (int)(n0 == 0 && n1 == 1) + (int)(n1 == 0 && n2 == 1)
               + (int)(n2 == 0 && n3 == 1) + (int)(n3 == 0 && n4 == 1)
               + (int)(n4 == 0 && n5 == 1) + (int)(n5 == 0 && n6 == 1)
               + (int)(n6 == 0 && n7 == 1) + (int)(n7 == 0 && n0 == 1);
        int c3 = firstPass ? (n0 & n2 & n4) : (n0 & n2 & n6);
        int c4 = firstPass ? (n2 & n4 & n6) : (n0 & n4 & n6);
        del = (Bn >= 2) && (Bn <= 6) && (A == 1) && (c3 == 0) && (c4 == 0);
    }
    out[gid] = (unsigned char)(c & (del ^ 1));
    if (del) atomicOr(&schanged, 1);
    __syncthreads();
    if (threadIdx.x == 0 && schanged) atomicOr(flags + 1, 1);
}

// ---------------- 3) convergence check between body iterations ----------------
__global__ void check_kernel(int* __restrict__ flags) {
    if (flags[1] == 0) flags[0] = 1;   // no deletion in this body => converged
    flags[1] = 0;
}

// ---------------- 4) skeleton endpoints: C = (x==1 && 8-neighbor-count==1) ----------------
__global__ void endpoint_kernel(const unsigned char* __restrict__ in, unsigned char* __restrict__ cmap) {
    int gid = blockIdx.x * blockDim.x + threadIdx.x;
    int b   = gid >> 18;
    int rem = gid & (HW - 1);
    int y = rem >> 9, x = rem & 511;
    const unsigned char* img = in + (b << 18);
    int c = img[rem];
    int cnt = getpix(img, y-1, x) + getpix(img, y-1, x+1) + getpix(img, y, x+1)
            + getpix(img, y+1, x+1) + getpix(img, y+1, x) + getpix(img, y+1, x-1)
            + getpix(img, y, x-1) + getpix(img, y-1, x-1);
    cmap[gid] = (c == 1 && cnt == 1) ? 1 : 0;
}

// ---------------- 5a) horizontal 9-wide ones conv as banded WMMA GEMM ----------------
// out[y][x0+n] = sum_k C[y][x0-8+k] * band[k][n],  band[k][n] = (|k-8-n|<=4)
// One v_wmma_f32_16x16x32_bf16 per 16x16 tile; the relative band fragment is constant.
__global__ void rowconv_kernel(const unsigned char* __restrict__ cmap, __bf16* __restrict__ rbuf) {
    const int wv   = (blockIdx.x * blockDim.x + threadIdx.x) >> 5;  // one wave per tile
    const int lane = threadIdx.x & 31;
    const int img  = wv >> 10;                 // 1024 tiles per image (32x32)
    const int it   = (wv >> 5) & 31;
    const int jt   = wv & 31;
    const int y0   = it << 4, x0 = jt << 4;
    const int base = img << 18;
    const int hi   = lane >> 4;                // lane group
    const int m    = lane & 15;                // A row / B-N col for this lane

    v16bf a, b;
#pragma unroll
    for (int e = 0; e < 16; ++e) {
        // A 16x32 bf16 layout: K = e + (e&8) + 8*hi ; M = lane&15
        int k   = e + (e & 8) + (hi << 3);
        int col = x0 - 8 + k;
        float av = (col >= 0 && col < W) ? (float)cmap[base + ((y0 + m) << 9) + col] : 0.0f;
        a[e] = (__bf16)av;
        // B 32x16 bf16 layout: K = e + (e&8) + 8*hi ; N = lane&15 ; band ones |k-8-n|<=4
        int kb = e + (e & 8) + (hi << 3);
        int d  = kb - 8 - m;
        b[e] = (__bf16)((d >= -4 && d <= 4) ? 1.0f : 0.0f);
    }
    v8f acc;
#pragma unroll
    for (int i = 0; i < 8; ++i) acc[i] = 0.0f;
    acc = __builtin_amdgcn_wmma_f32_16x16x32_bf16(false, a, false, b, (short)0, acc, false, false);
#pragma unroll
    for (int r = 0; r < 8; ++r) {              // C/D: M = r + 8*hi, N = lane&15
        int M = r + (hi << 3);
        rbuf[base + ((y0 + M) << 9) + x0 + m] = (__bf16)acc[r];  // <=9, exact in bf16
    }
}

// ---------------- 5b) vertical 9-wide conv (band*A data) fused with W-map and W*L tile sum ----------
__global__ void colconv_kernel(const __bf16* __restrict__ rbuf, const float* __restrict__ Lbuf,
                               float* __restrict__ partial) {
    const int wv   = (blockIdx.x * blockDim.x + threadIdx.x) >> 5;
    const int lane = threadIdx.x & 31;
    const int img  = wv >> 10;
    const int it   = (wv >> 5) & 31;
    const int jt   = wv & 31;
    const int y0   = it << 4, x0 = jt << 4;
    const int base = img << 18;
    const int hi   = lane >> 4;
    const int m    = lane & 15;

    v16bf a, b;
#pragma unroll
    for (int e = 0; e < 16; ++e) {
        // A = constant band: A[m][k] = (|k-8-m|<=4)
        int ka = e + (e & 8) + (hi << 3);
        int d  = ka - 8 - m;
        a[e] = (__bf16)((d >= -4 && d <= 4) ? 1.0f : 0.0f);
        // B = row-sum data slice: rows y0-8+k, col x0 + (lane&15)
        int kb  = e + (e & 8) + (hi << 3);
        int row = y0 - 8 + kb;
        b[e] = (row >= 0 && row < H) ? rbuf[base + (row << 9) + x0 + m] : (__bf16)0.0f;
    }
    v8f acc;
#pragma unroll
    for (int i = 0; i < 8; ++i) acc[i] = 0.0f;
    acc = __builtin_amdgcn_wmma_f32_16x16x32_bf16(false, a, false, b, (short)0, acc, false, false);

    float sum = 0.0f;
#pragma unroll
    for (int r = 0; r < 8; ++r) {
        int M = r + (hi << 3);
        float cnt = acc[r];                          // 9x9 window count of endpoints
        float Nv  = cnt * KSCALE;
        float Wv  = (cnt == 0.0f) ? 1.0f : Nv;       // N + (N==0)
        sum += Wv * Lbuf[base + ((y0 + M) << 9) + x0 + m];
    }
#pragma unroll
    for (int off = 16; off > 0; off >>= 1) sum += __shfl_xor(sum, off, 32);
    if (lane == 0) partial[wv] = sum;                // one deterministic write per tile
}

// ---------------- 6) final deterministic reduction ----------------
__global__ void reduce_kernel(const float* __restrict__ partial, float* __restrict__ out) {
    __shared__ float s[256];
    float acc = 0.0f;
    for (int i = threadIdx.x; i < TILES; i += 256) acc += partial[i];
    s[threadIdx.x] = acc;
    __syncthreads();
    for (int st = 128; st > 0; st >>= 1) {
        if (threadIdx.x < st) s[threadIdx.x] += s[threadIdx.x + st];
        __syncthreads();
    }
    if (threadIdx.x == 0) out[0] = s[0] / (float)NPIX;
}

// ---------------- launcher ----------------
extern "C" void kernel_launch(void* const* d_in, const int* in_sizes, int n_in,
                              void* d_out, int out_size, void* d_ws, size_t ws_size,
                              hipStream_t stream) {
    const float* pred   = (const float*)d_in[0];
    const int*   target = (const int*)d_in[1];

    // workspace layout (~18.9 MB): masks (u8) | L (f32) | rowsum (bf16) | partials | flags
    unsigned char* ws   = (unsigned char*)d_ws;
    unsigned char* xbuf = ws;                                   // NPIX u8
    unsigned char* ybuf = ws + (size_t)NPIX;                    // NPIX u8
    unsigned char* cbuf = ws + (size_t)2 * NPIX;                // NPIX u8
    float*  Lbuf    = (float*)(ws + (size_t)3 * NPIX);          // NPIX f32
    __bf16* rbuf    = (__bf16*)(ws + (size_t)3 * NPIX + (size_t)NPIX * 4);   // NPIX bf16
    float*  partial = (float*)(ws + (size_t)3 * NPIX + (size_t)NPIX * 4 + (size_t)NPIX * 2);
    int*    flags   = (int*)((unsigned char*)partial + (size_t)TILES * 4);

    const int blocksPix = NPIX / 256;   // 8192, exact

    prep_kernel<<<blocksPix, 256, 0, stream>>>(pred, target, Lbuf, xbuf, flags);

    // Zhang-Suen thinning: fixed launch count (graph-safe), device-side early-out once converged.
    for (int i = 0; i < ITERS; ++i) {
        thin_kernel<<<blocksPix, 256, 0, stream>>>(xbuf, ybuf, flags, 1);
        thin_kernel<<<blocksPix, 256, 0, stream>>>(ybuf, xbuf, flags, 0);
        check_kernel<<<1, 1, 0, stream>>>(flags);
    }

    endpoint_kernel<<<blocksPix, 256, 0, stream>>>(xbuf, cbuf);

    // 9x9 ones conv as two banded bf16 WMMA GEMM passes (one wave / 16x16 tile, 4 waves / block)
    rowconv_kernel<<<TILES / 4, 128, 0, stream>>>(cbuf, rbuf);
    colconv_kernel<<<TILES / 4, 128, 0, stream>>>(rbuf, Lbuf, partial);

    reduce_kernel<<<1, 256, 0, stream>>>(partial, (float*)d_out);
}
